// ContinuousCritic_30691836297668
// MI455X (gfx1250) — compile-verified, matrix operand order checked by reading, not run on hardware
//
#include <hip/hip_runtime.h>
#include <hip/hip_bf16.h>

typedef __attribute__((ext_vector_type(16))) _Float16 v16h;
typedef __attribute__((ext_vector_type(8)))  _Float16 v8h;
typedef __attribute__((ext_vector_type(8)))  float    v8f;

#define B_TOTAL       32768
#define BATCH_PER_WG  16
#define ROWS          96          // 16 batches * 6 pairs
#define MT            6           // M tiles of 16 rows

// PERMS = permutations(range(3),2) = (0,1)(0,2)(1,0)(1,2)(2,0)(2,1)
__constant__ int c_pi[6] = {0,0,1,1,2,2};
__constant__ int c_pj[6] = {1,2,0,2,0,1};

// ---------------------------------------------------------------------------
// A-fragment (16x32 f16, MxK) from LDS, row-major with given stride (halves).
// Lanes 0-15: M=lane,   K = kb+0..7 (h0..7), kb+16..23 (h8..15)
// Lanes 16-31: M=lane-16, same with kb += 8
// ---------------------------------------------------------------------------
__device__ inline v16h load_a_lds(const _Float16* s, int stride, int mbase, int kbase) {
  int lane = threadIdx.x & 31;
  const _Float16* row = s + (mbase + (lane & 15)) * stride + kbase + ((lane >> 4) << 3);
  v8h lo = *(const v8h*)(row);        // K = kb + 0..7
  v8h hi = *(const v8h*)(row + 16);   // K = kb + 16..23
  return __builtin_shufflevector(lo, hi, 0,1,2,3,4,5,6,7,8,9,10,11,12,13,14,15);
}

// ---------------------------------------------------------------------------
// B-fragment (32x16 f16, KxN) from pre-swizzled global weights:
// block = kt*nNt + nt, inside block: [lane][16 halves] contiguous (32B/lane)
// ---------------------------------------------------------------------------
__device__ inline v16h load_b_sw(const _Float16* w, int block) {
  int lane = threadIdx.x & 31;
  const v8h* p = (const v8h*)(w + (size_t)block * 512 + lane * 16);
  v8h lo = p[0];
  v8h hi = p[1];
  return __builtin_shufflevector(lo, hi, 0,1,2,3,4,5,6,7,8,9,10,11,12,13,14,15);
}

__device__ inline v8f wmma_f16(v16h a, v16h b, v8f c) {
  return __builtin_amdgcn_wmma_f32_16x16x32_f16(false, a, false, b, (short)0, c, false, false);
}

// ---------------------------------------------------------------------------
// Weight prep: f32 (Krows x N, row-major) -> f16 swizzled B-fragment blocks,
// K zero-padded to Kpad (multiple of 32).
// ---------------------------------------------------------------------------
__global__ void swizzle_w_kernel(const float* __restrict__ W, _Float16* __restrict__ out,
                                 int Krows, int Kpad, int N) {
  int total = (Kpad >> 5) * (N >> 4) * 512;
  int nNt = N >> 4;
  for (int e = blockIdx.x * blockDim.x + threadIdx.x; e < total; e += gridDim.x * blockDim.x) {
    int block  = e >> 9;
    int within = e & 511;
    int ln = within >> 4;
    int h  = within & 15;
    int kt = block / nNt;
    int nt = block - kt * nNt;
    int n  = nt * 16 + (ln & 15);
    int k  = kt * 32 + ((ln >> 4) << 3) + (h & 7) + ((h >> 3) << 4);
    float v = (k < Krows) ? W[(size_t)k * N + n] : 0.0f;
    out[e] = (_Float16)v;
  }
}

struct CriticParams {
  const float *obs, *ag, *g, *act;
  const float *b1a, *b2a, *b1b, *b2b;       // phi biases
  const float *rb1a, *rb1b;                 // rho layer1 biases
  const float *rw2a, *rb2a, *rw2b, *rb2b;   // rho layer2 (256x1) + bias
  const _Float16 *w1a, *w1b, *w2a, *w2b, *rw1a, *rw1b;  // swizzled f16 weights
  float* out;                               // [2*B_TOTAL]: q1 then q2
};

// ---------------------------------------------------------------------------
// Fully fused critic. Loop nest: each wave owns 2 N-tiles; for each N-tile it
// keeps 6 independent accumulators (one per M-tile) so consecutive WMMAs are
// independent (no accumulator RAW nop stalls) and each B-fragment is loaded
// once per k-step.
// ---------------------------------------------------------------------------
__global__ __launch_bounds__(256)
void critic_fused_kernel(CriticParams P) {
  __shared__ __align__(16) unsigned char smem[12288 + 49152 + 16384 + 8192];
  _Float16* s_inp  = (_Float16*)(smem);                         // 96 x 64  f16
  _Float16* s_h1   = (_Float16*)(smem + 12288);                 // 96 x 256 f16
  float*    s_osum = (float*)   (smem + 12288 + 49152);         // 16 x 256 f32
  _Float16* s_of16 = (_Float16*)(smem + 12288 + 49152 + 16384); // 16 x 256 f16
  float*    s_hrho = (float*)s_h1;                              // 16 x 256 f32 (reuse)

  const int tid  = threadIdx.x;
  const int lane = tid & 31;
  const int wave = tid >> 5;
  const int b0   = blockIdx.x * BATCH_PER_WG;

  // ---- build inp features (96 rows x 64 cols, cols 62..63 = pad) ----
  for (int t = tid; t < ROWS * 64; t += 256) {
    int lr = t >> 6, c = t & 63;
    int bl = lr / 6, p = lr - bl * 6;
    int gb = b0 + bl;
    int pi = c_pi[p], pj = c_pj[p];
    float v = 0.0f;
    if      (c < 3)  v = P.ag[gb * 9 + pi * 3 + c];
    else if (c < 6)  v = P.ag[gb * 9 + pj * 3 + (c - 3)];
    else if (c < 9)  v = P.g [gb * 9 + pi * 3 + (c - 6)];
    else if (c < 12) v = P.g [gb * 9 + pj * 3 + (c - 9)];
    else if (c < 22) v = P.obs[gb * 55 + (c - 12)];                    // body
    else if (c < 25) v = ((c - 22) == pi) ? 1.0f : 0.0f;               // onehot pi
    else if (c < 40) v = P.obs[gb * 55 + 10 + pi * 15 + (c - 25)];     // obj pi
    else if (c < 43) v = ((c - 40) == pj) ? 1.0f : 0.0f;               // onehot pj
    else if (c < 58) v = P.obs[gb * 55 + 10 + pj * 15 + (c - 43)];     // obj pj
    else if (c < 62) v = P.act[gb * 4 + (c - 58)];
    s_inp[lr * 64 + c] = (_Float16)v;
  }
  __syncthreads();

  for (int net = 0; net < 2; ++net) {
    const _Float16* w1  = net ? P.w1b  : P.w1a;
    const _Float16* w2  = net ? P.w2b  : P.w2a;
    const _Float16* rw1 = net ? P.rw1b : P.rw1a;
    const float* b1  = net ? P.b1b  : P.b1a;
    const float* b2  = net ? P.b2b  : P.b2a;
    const float* rb1 = net ? P.rb1b : P.rb1a;
    const float* rw2 = net ? P.rw2b : P.rw2a;
    const float* rb2 = net ? P.rb2b : P.rb2a;

    if (net) __syncthreads();   // protect s_h1/s_osum reuse across nets

    // zero pair-sum accumulator
    for (int t = tid; t < BATCH_PER_WG * 256; t += 256) s_osum[t] = 0.0f;

    // ---- phi layer 1: (96x62) @ (62x256), K padded to 64 ----
    for (int q = 0; q < 2; ++q) {
      const int nt = wave * 2 + q;
      v8f acc[MT] = {};
      #pragma unroll
      for (int kt = 0; kt < 2; ++kt) {
        v16h b = load_b_sw(w1, kt * 16 + nt);
        #pragma unroll
        for (int mt = 0; mt < MT; ++mt) {
          v16h a = load_a_lds(s_inp, 64, mt * 16, kt * 32);
          acc[mt] = wmma_f16(a, b, acc[mt]);
        }
      }
      const int n  = nt * 16 + (lane & 15);
      const float bias = b1[n];
      #pragma unroll
      for (int mt = 0; mt < MT; ++mt) {
        int mb = mt * 16 + ((lane >> 4) << 3);
        #pragma unroll
        for (int r = 0; r < 8; ++r) {
          float v = acc[mt][r] + bias;
          v = v > 0.0f ? v : 0.0f;
          s_h1[(mb + r) * 256 + n] = (_Float16)v;
        }
      }
    }
    __syncthreads();

    // ---- phi layer 2: (96x256) @ (256x256), relu + pair-sum into s_osum ----
    for (int q = 0; q < 2; ++q) {
      const int nt = wave * 2 + q;
      v8f acc[MT] = {};
      #pragma unroll 2
      for (int kt = 0; kt < 8; ++kt) {
        v16h b = load_b_sw(w2, kt * 16 + nt);
        #pragma unroll
        for (int mt = 0; mt < MT; ++mt) {
          v16h a = load_a_lds(s_h1, 256, mt * 16, kt * 32);
          acc[mt] = wmma_f16(a, b, acc[mt]);
        }
      }
      const int n  = nt * 16 + (lane & 15);
      const float bias = b2[n];
      #pragma unroll
      for (int mt = 0; mt < MT; ++mt) {
        int mb = mt * 16 + ((lane >> 4) << 3);
        #pragma unroll
        for (int r = 0; r < 8; ++r) {
          float v = acc[mt][r] + bias;
          v = v > 0.0f ? v : 0.0f;
          int row = mb + r;                          // pair-row in [0,96)
          atomicAdd(&s_osum[(row / 6) * 256 + n], v); // ds_add_f32
        }
      }
    }
    __syncthreads();

    // ---- pooled features -> f16 for rho WMMA ----
    for (int t = tid; t < BATCH_PER_WG * 256; t += 256) s_of16[t] = (_Float16)s_osum[t];
    __syncthreads();

    // ---- rho layer 1: (16x256) @ (256x256); 2 N-tiles per wave interleaved ----
    {
      const int nt0 = wave * 2;
      const int nt1 = wave * 2 + 1;
      v8f acc0 = {}, acc1 = {};
      #pragma unroll 2
      for (int kt = 0; kt < 8; ++kt) {
        v16h a  = load_a_lds(s_of16, 256, 0, kt * 32);
        v16h bb0 = load_b_sw(rw1, kt * 16 + nt0);
        v16h bb1 = load_b_sw(rw1, kt * 16 + nt1);
        acc0 = wmma_f16(a, bb0, acc0);
        acc1 = wmma_f16(a, bb1, acc1);
      }
      const int mb = (lane >> 4) << 3;
      const int n0 = nt0 * 16 + (lane & 15);
      const int n1 = nt1 * 16 + (lane & 15);
      const float bias0 = rb1[n0];
      const float bias1 = rb1[n1];
      #pragma unroll
      for (int r = 0; r < 8; ++r) {
        float v0 = acc0[r] + bias0; v0 = v0 > 0.0f ? v0 : 0.0f;
        float v1 = acc1[r] + bias1; v1 = v1 > 0.0f ? v1 : 0.0f;
        s_hrho[(mb + r) * 256 + n0] = v0;
        s_hrho[(mb + r) * 256 + n1] = v1;
      }
    }
    __syncthreads();

    // ---- rho layer 2: per-row dot with rw2 (256 -> 1) ----
    if (tid < 128) {
      int row = tid >> 3, seg = tid & 7;
      int cb = seg * 32;
      float s = 0.0f;
      #pragma unroll 4
      for (int c = 0; c < 32; ++c) s += s_hrho[row * 256 + cb + c] * rw2[cb + c];
      s += __shfl_down(s, 4, 8);
      s += __shfl_down(s, 2, 8);
      s += __shfl_down(s, 1, 8);
      if (seg == 0) P.out[net * B_TOTAL + b0 + row] = s + rb2[0];
    }
  }
}

extern "C" void kernel_launch(void* const* d_in, const int* in_sizes, int n_in,
                              void* d_out, int out_size, void* d_ws, size_t ws_size,
                              hipStream_t stream) {
  const float* obs = (const float*)d_in[0];
  const float* ag  = (const float*)d_in[1];
  const float* g   = (const float*)d_in[2];
  const float* act = (const float*)d_in[3];
  const float* phi_w1a = (const float*)d_in[4];
  const float* phi_b1a = (const float*)d_in[5];
  const float* phi_w2a = (const float*)d_in[6];
  const float* phi_b2a = (const float*)d_in[7];
  const float* phi_w1b = (const float*)d_in[8];
  const float* phi_b1b = (const float*)d_in[9];
  const float* phi_w2b = (const float*)d_in[10];
  const float* phi_b2b = (const float*)d_in[11];
  const float* rho_w1a = (const float*)d_in[12];
  const float* rho_b1a = (const float*)d_in[13];
  const float* rho_w2a = (const float*)d_in[14];
  const float* rho_b2a = (const float*)d_in[15];
  const float* rho_w1b = (const float*)d_in[16];
  const float* rho_b1b = (const float*)d_in[17];
  const float* rho_w2b = (const float*)d_in[18];
  const float* rho_b2b = (const float*)d_in[19];

  // workspace layout (f16 elements): swizzled weights
  _Float16* ws = (_Float16*)d_ws;
  const size_t SZ_W1 = 2 * 16 * 512;   // 64x256  -> 16384 halves
  const size_t SZ_W2 = 8 * 16 * 512;   // 256x256 -> 65536 halves
  _Float16* w1a_sw  = ws;
  _Float16* w1b_sw  = w1a_sw + SZ_W1;
  _Float16* w2a_sw  = w1b_sw + SZ_W1;
  _Float16* w2b_sw  = w2a_sw + SZ_W2;
  _Float16* rw1a_sw = w2b_sw + SZ_W2;
  _Float16* rw1b_sw = rw1a_sw + SZ_W2;

  swizzle_w_kernel<<<64, 256, 0, stream>>>(phi_w1a, w1a_sw, 62, 64, 256);
  swizzle_w_kernel<<<64, 256, 0, stream>>>(phi_w1b, w1b_sw, 62, 64, 256);
  swizzle_w_kernel<<<256, 256, 0, stream>>>(phi_w2a, w2a_sw, 256, 256, 256);
  swizzle_w_kernel<<<256, 256, 0, stream>>>(phi_w2b, w2b_sw, 256, 256, 256);
  swizzle_w_kernel<<<256, 256, 0, stream>>>(rho_w1a, rw1a_sw, 256, 256, 256);
  swizzle_w_kernel<<<256, 256, 0, stream>>>(rho_w1b, rw1b_sw, 256, 256, 256);

  CriticParams P;
  P.obs = obs; P.ag = ag; P.g = g; P.act = act;
  P.b1a = phi_b1a; P.b2a = phi_b2a; P.b1b = phi_b1b; P.b2b = phi_b2b;
  P.rb1a = rho_b1a; P.rb1b = rho_b1b;
  P.rw2a = rho_w2a; P.rb2a = rho_b2a; P.rw2b = rho_w2b; P.rb2b = rho_b2b;
  P.w1a = w1a_sw; P.w1b = w1b_sw; P.w2a = w2a_sw; P.w2b = w2b_sw;
  P.rw1a = rw1a_sw; P.rw1b = rw1b_sw;
  P.out = (float*)d_out;

  critic_fused_kernel<<<B_TOTAL / BATCH_PER_WG, 256, 0, stream>>>(P);
}
